// AttentionLayer_65575560675684
// MI455X (gfx1250) — matrix operand order repaired and checked by reading, not compiled, not run on hardware
//
#include <hip/hip_runtime.h>
#include <float.h>

typedef __attribute__((ext_vector_type(16))) _Float16 v16h;
typedef __attribute__((ext_vector_type(8)))  _Float16 v8h;
typedef __attribute__((ext_vector_type(4)))  _Float16 v4h;
typedef __attribute__((ext_vector_type(2)))  _Float16 v2h;
typedef __attribute__((ext_vector_type(8)))  float    v8f;
typedef __attribute__((ext_vector_type(4)))  float    v4f;

#define GN 8192
#define GD 256
#define MT 32                 // output rows per block in k_spmm
#define KT 32                 // K per WMMA step
#define NSTEP (GN / KT)       // 256

// ---------------- K1: scores[j] = inputs[j,:] . H_v ----------------
__global__ __launch_bounds__(256) void k_scores(const float* __restrict__ x,
                                                const float* __restrict__ hv,
                                                float* __restrict__ sc) {
  const int wave = threadIdx.x >> 5, lane = threadIdx.x & 31;
  const int row = (blockIdx.x << 3) + wave;
  const float* rp = x + (size_t)row * GD;
  float acc = 0.f;
#pragma unroll
  for (int i = 0; i < GD; i += 32) acc = fmaf(rp[i + lane], hv[i + lane], acc);
#pragma unroll
  for (int off = 16; off; off >>= 1) acc += __shfl_xor(acc, off, 32);
  if (!lane) sc[row] = acc;
}

// ---------------- K1.5: xh = (f16) inputs  (one-time convert) ----------------
__global__ __launch_bounds__(256) void k_half(const float* __restrict__ x,
                                              _Float16* __restrict__ xh) {
  const int i = (blockIdx.x * 256 + threadIdx.x) << 2;
  v4f v = *(const v4f*)(x + i);
  v4h h;
#pragma unroll
  for (int q = 0; q < 4; ++q) h[q] = (_Float16)v[q];
  *(v4h*)(xh + i) = h;
}

// ------- K2: per-row online softmax stats over nonzeros of adj -------
__global__ __launch_bounds__(256) void k_rowstats(const float* __restrict__ adj,
                                                  const float* __restrict__ sc,
                                                  float* __restrict__ rmax,
                                                  float* __restrict__ rinv) {
  __shared__ float ssc[GN];                              // 32 KB staged scores
  for (int i = threadIdx.x; i < GN; i += 256) ssc[i] = sc[i];
  __syncthreads();
  const int wave = threadIdx.x >> 5, lane = threadIdx.x & 31;
  const int row = (blockIdx.x << 3) + wave;
  const float* rp = adj + (size_t)row * GN;
  float m = -FLT_MAX, s = 0.f;
  for (int j = 0; j < GN; j += 128) {                    // wave covers 128 cols/iter
    v4f a = __builtin_nontemporal_load((const v4f*)(rp + j + (lane << 2)));
    v4f v = *(const v4f*)(ssc + j + (lane << 2));
#pragma unroll
    for (int q = 0; q < 4; ++q) {
      if (a[q] != 0.f) {
        float l = a[q] * v[q];
        float nm = fmaxf(m, l);
        s = s * __expf(m - nm) + __expf(l - nm);
        m = nm;
      }
    }
  }
#pragma unroll
  for (int off = 16; off; off >>= 1) {                   // merge (m,s) pairs
    float om = __shfl_xor(m, off, 32);
    float os = __shfl_xor(s, off, 32);
    float nm = fmaxf(m, om);
    s = s * __expf(m - nm) + os * __expf(om - nm);
    m = nm;
  }
  if (!lane) { rmax[row] = m; rinv[row] = (s > 0.f) ? (1.f / s) : 0.f; }
}

// ------- K3: out = softmax_weights(adj) @ inputs, via f16 WMMA -------
// Fragment-ready LDS layouts: each lane's 16-half WMMA operand is one
// contiguous 32B block (2x ds_load_b128).
//   sA half index: mtile*512 + h*256 + (row&15)*16 + p
//   sB half index: ntile*512 + (k>>4)*256 + (col&15)*16 + (k&15)
// MODE==1: B staged from pre-converted f16 copy (b128 loads).
// MODE==0: B staged from f32 inputs (fallback when ws too small).
template <int MODE>
__global__ __launch_bounds__(256) void k_spmm(const float* __restrict__ adj,
                                              const float* __restrict__ x,
                                              const _Float16* __restrict__ xh,
                                              const float* __restrict__ sc,
                                              const float* __restrict__ rmax,
                                              const float* __restrict__ rinv,
                                              float* __restrict__ out) {
  __shared__ alignas(32) _Float16 sA[2][2 * 512];        // 2 x 2KB  (2 M-tiles)
  __shared__ alignas(32) _Float16 sB[2][16 * 512];       // 2 x 16KB (16 N-tiles)

  const int t = threadIdx.x;
  const int lane = t & 31, wv = t >> 5;                  // 8 waves
  const int row0 = blockIdx.x * MT;

  // A-staging constants: thread handles row ar, k-elements kk0..kk0+3
  const int ar  = t >> 3;                                // 0..31
  const int kk0 = (t & 7) << 2;                          // 0,4,...,28
  const int ah  = kk0 >> 4;
  const int ap  = kk0 & 15;
  const int aoff = ((ar >> 4) << 9) + (ah << 8) + ((ar & 15) << 4) + ap;
  const int arow = row0 + ar;
  const float* adjp = adj + (size_t)arow * GN + kk0;
  const float mval = rmax[arow];
  const float ival = rinv[arow];

  // B-staging constants
  const int bc = t;                                      // MODE0: column 0..255
  const int boffc = ((bc >> 4) << 9) + ((bc & 15) << 4);

  // staging registers (live across the barrier)
  v4f La = {}, Ls = {};
  v8h Lb00 = {}, Lb01 = {}, Lb10 = {}, Lb11 = {};

  v8f acc00 = {}, acc01 = {}, acc10 = {}, acc11 = {};

#define A_LOAD(KB)                                                              \
  {                                                                             \
    La = __builtin_nontemporal_load((const v4f*)(adjp + (KB)));                 \
    Ls = *(const v4f*)(sc + (KB) + kk0);                                        \
  }
#define A_STORE(BUF)                                                            \
  {                                                                             \
    v4h w;                                                                      \
    _Pragma("unroll")                                                           \
    for (int q = 0; q < 4; ++q)                                                 \
      w[q] = (La[q] != 0.f)                                                     \
               ? (_Float16)(__expf(fmaf(La[q], Ls[q], -mval)) * ival)           \
               : (_Float16)0.f;                                                 \
    *(v4h*)(&sA[(BUF)][aoff]) = w;                                              \
  }
// MODE1: 2 tasks/thread; task = (k-pair kp, 8-col group cg); 2x b128 each
#define B_LOAD1(KB)                                                             \
  {                                                                             \
    const int kp0 = t >> 5, cg0 = t & 31;                                       \
    const _Float16* p0 = xh + (size_t)((KB) + (kp0 << 1)) * GD + (cg0 << 3);    \
    Lb00 = *(const v8h*)p0;                                                     \
    Lb01 = *(const v8h*)(p0 + GD);                                              \
    const int kp1 = kp0 + 8;                                                    \
    const _Float16* p1 = xh + (size_t)((KB) + (kp1 << 1)) * GD + (cg0 << 3);    \
    Lb10 = *(const v8h*)p1;                                                     \
    Lb11 = *(const v8h*)(p1 + GD);                                              \
  }
#define B_STORE1(BUF)                                                           \
  {                                                                             \
    const int cg0 = t & 31;                                                     \
    const int kpA = t >> 5, kkA = kpA << 1, grpA = kpA >> 3;                    \
    const int kpB = kpA + 8, kkB = kpB << 1, grpB = kpB >> 3;                   \
    _Pragma("unroll")                                                           \
    for (int q = 0; q < 8; ++q) {                                               \
      const int c = (cg0 << 3) + q;                                             \
      const int cb = ((c >> 4) << 9) + ((c & 15) << 4);                         \
      v2h pA; pA[0] = Lb00[q]; pA[1] = Lb01[q];                                 \
      *(v2h*)(&sB[(BUF)][cb + (grpA << 8) + (kkA & 15)]) = pA;                  \
      v2h pB; pB[0] = Lb10[q]; pB[1] = Lb11[q];                                 \
      *(v2h*)(&sB[(BUF)][cb + (grpB << 8) + (kkB & 15)]) = pB;                  \
    }                                                                           \
  }
// MODE0 fallback: column-per-thread from f32 inputs
#define B_STAGE0(BUF, KB)                                                       \
  {                                                                             \
    _Pragma("unroll")                                                           \
    for (int kp = 0; kp < 16; ++kp) {                                           \
      const int kk = kp << 1;                                                   \
      float x0 = x[(size_t)((KB) + kk) * GD + bc];                              \
      float x1 = x[(size_t)((KB) + kk + 1) * GD + bc];                          \
      v2h p; p[0] = (_Float16)x0; p[1] = (_Float16)x1;                          \
      *(v2h*)(&sB[(BUF)][boffc + ((kk >> 4) << 8) + (kk & 15)]) = p;            \
    }                                                                           \
  }

  // prologue: stage step 0 (no hazard yet)
  A_LOAD(0) A_STORE(0)
  if constexpr (MODE == 1) { B_LOAD1(0) B_STORE1(0) }
  else                     { B_STAGE0(0, 0) }

  const int lh = lane >> 4, ll = lane & 15;
  for (int step = 0; step < NSTEP; ++step) {
    const int cur = step & 1;
    const bool more = (step + 1 < NSTEP);
    const int kb = (step + 1) * KT;
    if (more) {                                  // global loads before barrier
      A_LOAD(kb)
      if constexpr (MODE == 1) B_LOAD1(kb)
    }
    __syncthreads();
    if (more) {                                  // LDS stores after barrier
      A_STORE(cur ^ 1)
      if constexpr (MODE == 1) B_STORE1(cur ^ 1)
      else                     B_STAGE0(cur ^ 1, kb)
    }

    v16h a0 = *(const v16h*)&sA[cur][(0 << 9) + (lh << 8) + (ll << 4)];
    v16h a1 = *(const v16h*)&sA[cur][(1 << 9) + (lh << 8) + (ll << 4)];
    v16h b0 = *(const v16h*)&sB[cur][(wv << 9) + (lh << 8) + (ll << 4)];
    v16h b1 = *(const v16h*)&sB[cur][((wv + 8) << 9) + (lh << 8) + (ll << 4)];

    acc00 = __builtin_amdgcn_wmma_f32_16x16x32_f16(false, a0, false, b0,
                                                   (short)0, acc00, false, false);
    acc01 = __builtin_amdgcn_wmma_f32_16x16x32_f16(false, a0, false, b1,
                                                   (short)0, acc01, false, false);
    acc10 = __builtin_amdgcn_wmma_f32_16x16x32_f16(false, a1, false, b0,
                                                   (short)0, acc10, false, false);
    acc11 = __builtin_amdgcn_wmma_f32_16x16x32_f16(false, a1, false, b1,
                                                   (short)0, acc11, false, false);
  }
#undef A_LOAD
#undef A_STORE
#undef B_LOAD1
#undef B_STORE1
#undef B_STAGE0

  // C/D layout: VGPR a -> M = a + 8*(lane>=16), N = lane&15
#pragma unroll
  for (int a = 0; a < 8; ++a) {
    const int r0 = row0 + a + lh * 8;
    const int r1 = r0 + 16;
    out[(size_t)r0 * GD + (wv << 4) + ll]        = acc00[a];
    out[(size_t)r0 * GD + ((wv + 8) << 4) + ll]  = acc01[a];
    out[(size_t)r1 * GD + (wv << 4) + ll]        = acc10[a];
    out[(size_t)r1 * GD + ((wv + 8) << 4) + ll]  = acc11[a];
  }
}

extern "C" void kernel_launch(void* const* d_in, const int* in_sizes, int n_in,
                              void* d_out, int out_size, void* d_ws, size_t ws_size,
                              hipStream_t stream) {
  (void)in_sizes; (void)n_in; (void)out_size;
  const float* inputs = (const float*)d_in[0];   // [8192, 256]
  const float* adj    = (const float*)d_in[1];   // [8192, 8192]
  const float* hv     = (const float*)d_in[2];   // [256, 1]
  float* out  = (float*)d_out;                   // [8192, 256]
  float* sc   = (float*)d_ws;                    // scores [8192]
  float* rmx  = sc + GN;                         // row max [8192]
  float* rin  = rmx + GN;                        // 1/row-sum [8192]
  _Float16* xh = (_Float16*)(rin + GN);          // f16 inputs copy [8192*256]

  const size_t need = (size_t)3 * GN * sizeof(float) +
                      (size_t)GN * GD * sizeof(_Float16);

  k_scores  <<<GN / 8, 256, 0, stream>>>(inputs, hv, sc);
  k_rowstats<<<GN / 8, 256, 0, stream>>>(adj, sc, rmx, rin);
  if (ws_size >= need) {
    k_half   <<<(GN * GD) / 1024, 256, 0, stream>>>(inputs, xh);
    k_spmm<1><<<GN / MT, 256, 0, stream>>>(adj, inputs, xh, sc, rmx, rin, out);
  } else {
    k_spmm<0><<<GN / MT, 256, 0, stream>>>(adj, inputs, xh, sc, rmx, rin, out);
  }
}
